// LocalSelfAttention_72224170049618
// MI455X (gfx1250) — matrix operand order repaired and checked by reading, not compiled, not run on hardware
//
#include <hip/hip_runtime.h>
#include <hip/hip_bf16.h>

typedef __attribute__((ext_vector_type(16))) _Float16 v16h;
typedef __attribute__((ext_vector_type(8)))  _Float16 v8h;
typedef __attribute__((ext_vector_type(8)))  float    v8f;

#define SLEN   4096
#define BATCH  2
#define DMODEL 768
#define NH     12
#define HDIM   64
#define WSZ    256
#define MROWS  (SLEN * BATCH)   // 8192

#if defined(__HIP_DEVICE_COMPILE__) && __has_builtin(__builtin_amdgcn_tensor_load_to_lds) && __has_builtin(__builtin_amdgcn_s_wait_tensorcnt)
#define HAVE_TDM 1
#else
#define HAVE_TDM 0
#endif

// Load a 16x32 f16 A-fragment (contiguous K layout, matching B) from a row base
// pointer with row stride already applied. lane<16 gets K{0..15},
// lane>=16 gets K{16..31}: one contiguous 32-byte chunk.
__device__ __forceinline__ v16h load_a_frag(const _Float16* base, int half) {
    return *(const v16h*)(base + 16 * half);
}

#if HAVE_TDM
typedef __attribute__((ext_vector_type(4))) unsigned int u32x4;
typedef __attribute__((ext_vector_type(8))) int          i32x8;
typedef __attribute__((ext_vector_type(4))) int          i32x4;

// Issue one TDM 2D tile load: f16 tensor (tdim0 x tdim1, row stride stride0
// elements), tile (tile_k x tile_rows) starting at gsrc, packed into LDS at
// lds_off. D# bit layout per CDNA5 ISA 8.3/8.4 (data_size=1 -> 2 bytes,
// type=2, count=1, groups 2/3 zero for 2D tiles). 6-arg builtin form
// (clang-23 / therock-10.0 headers).
__device__ __forceinline__ void tdm_load_2d(const _Float16* gsrc, unsigned lds_off,
                                            unsigned tile_k, unsigned tile_rows,
                                            unsigned tdim0, unsigned tdim1,
                                            unsigned stride0) {
    unsigned long long ga = (unsigned long long)(uintptr_t)gsrc;
    u32x4 g0;
    g0[0] = 1u;                                                  // count=1
    g0[1] = lds_off;                                             // lds_addr
    g0[2] = (unsigned)ga;                                        // global_addr lo
    g0[3] = ((unsigned)(ga >> 32) & 0x01FFFFFFu) | 0x80000000u;  // addr hi | type=2
    i32x8 g1;
    g1[0] = (int)(1u << 16);                                     // data_size=2B
    g1[1] = (int)(tdim0 << 16);                                  // tensor_dim0 lo
    g1[2] = (int)((tdim0 >> 16) | (tdim1 << 16));                // dim0 hi | dim1 lo
    g1[3] = (int)((tdim1 >> 16) | (tile_k << 16));               // dim1 hi | tile_dim0
    g1[4] = (int)tile_rows;                                      // tile_dim1 (dim2=0)
    g1[5] = (int)stride0;                                        // dim0_stride lo
    g1[6] = 0;
    g1[7] = 0;
    i32x4 gz4 = {0, 0, 0, 0};
    i32x8 gz8 = {0, 0, 0, 0, 0, 0, 0, 0};
    __builtin_amdgcn_tensor_load_to_lds(g0, g1, gz4, gz4, gz8, 0);
}
#endif

// ---------------------------------------------------------------------------
// Prep kernel A: pack val f32 -> f16 (row-major [m][k], m = s*B+b)
// ---------------------------------------------------------------------------
__global__ __launch_bounds__(256)
void pack_val(const float* __restrict__ val, _Float16* __restrict__ val_h) {
    const size_t idx = (size_t)blockIdx.x * 256 + threadIdx.x;  // one float4 each
    float4 v = *(const float4*)(val + idx * 4);
    union { _Float16 h[4]; uint2 u; } p;
    p.h[0] = (_Float16)v.x; p.h[1] = (_Float16)v.y;
    p.h[2] = (_Float16)v.z; p.h[3] = (_Float16)v.w;
    ((uint2*)val_h)[idx] = p.u;
}

// ---------------------------------------------------------------------------
// Prep kernel B: transpose W (768x768, [k][n]) -> Wt f16 ([z][n][k])
// ---------------------------------------------------------------------------
__global__ __launch_bounds__(256)
void transpose_w(const float* __restrict__ Wq, const float* __restrict__ Wk,
                 const float* __restrict__ Wv, _Float16* __restrict__ Wt) {
    __shared__ _Float16 t[32][33];
    const int z = blockIdx.z;
    const float* W = (z == 0) ? Wq : (z == 1) ? Wk : Wv;
    _Float16* dst = Wt + (size_t)z * DMODEL * DMODEL;
    const int n0 = blockIdx.x * 32, k0 = blockIdx.y * 32;
    const int tx = threadIdx.x & 31, ty = threadIdx.x >> 5;  // 32 x 8
    #pragma unroll
    for (int i = 0; i < 4; ++i) {
        const int k = ty + i * 8;
        t[k][tx] = (_Float16)W[(size_t)(k0 + k) * DMODEL + n0 + tx];
    }
    __syncthreads();
    #pragma unroll
    for (int i = 0; i < 4; ++i) {
        const int n = ty + i * 8;
        dst[(size_t)(n0 + n) * DMODEL + k0 + tx] = t[tx][n];
    }
}

// ---------------------------------------------------------------------------
// Kernel 1: QKV projection GEMM, f16 in / f32 accumulate.
// Block = 256 threads (8 waves) -> 128x64 tile; BK=32. Tiles staged by the
// Tensor Data Mover (double-buffered, wave 0 issues, TENSORcnt-gated).
// ---------------------------------------------------------------------------
__global__ __launch_bounds__(256)
void qkv_gemm(const _Float16* __restrict__ val_h, const _Float16* __restrict__ Wt,
              const float* __restrict__ bq, const float* __restrict__ bk,
              const float* __restrict__ bv,
              _Float16* __restrict__ qf, _Float16* __restrict__ kf,
              _Float16* __restrict__ vt)
{
    const int tid  = threadIdx.x;
    const int lane = tid & 31;
    const int wave = tid >> 5;
    const int half = lane >> 4;
    const int r    = lane & 15;

    const int n0 = blockIdx.x * 64;
    const int m0 = blockIdx.y * 128;
    const int z  = blockIdx.z;

    const _Float16* Wz = Wt + (size_t)z * DMODEL * DMODEL;
    const float* bias  = (z == 0) ? bq : (z == 1) ? bk : bv;

    v8f acc[4];
    #pragma unroll
    for (int nt = 0; nt < 4; ++nt)
        #pragma unroll
        for (int e = 0; e < 8; ++e) acc[nt][e] = 0.0f;

#if HAVE_TDM
    __shared__ _Float16 As[2][128 * 32];  // [m][k]
    __shared__ _Float16 Bs[2][64 * 32];   // [n][k]
    const int NSTEP = DMODEL / 32;
    if (wave == 0) {
        tdm_load_2d(val_h + (size_t)m0 * DMODEL, (unsigned)(uintptr_t)&As[0][0],
                    32, 128, DMODEL, MROWS, DMODEL);
        tdm_load_2d(Wz + (size_t)n0 * DMODEL, (unsigned)(uintptr_t)&Bs[0][0],
                    32, 64, DMODEL, DMODEL, DMODEL);
    }
    for (int it = 0; it < NSTEP; ++it) {
        const int cur = it & 1;
        if (wave == 0) {
            if (it + 1 < NSTEP) {
                const int kk = (it + 1) * 32;
                tdm_load_2d(val_h + (size_t)m0 * DMODEL + kk,
                            (unsigned)(uintptr_t)&As[cur ^ 1][0],
                            32, 128, DMODEL, MROWS, DMODEL);
                tdm_load_2d(Wz + (size_t)n0 * DMODEL + kk,
                            (unsigned)(uintptr_t)&Bs[cur ^ 1][0],
                            32, 64, DMODEL, DMODEL, DMODEL);
                __builtin_amdgcn_s_wait_tensorcnt(2);  // current pair complete
            } else {
                __builtin_amdgcn_s_wait_tensorcnt(0);
            }
        }
        __syncthreads();
        const v16h a = load_a_frag(&As[cur][(wave * 16 + r) * 32], half);
        #pragma unroll
        for (int nt = 0; nt < 4; ++nt) {
            const v16h b = *(const v16h*)&Bs[cur][(nt * 16 + r) * 32 + 16 * half];
            acc[nt] = __builtin_amdgcn_wmma_f32_16x16x32_f16(
                false, a, false, b, (short)0, acc[nt], false, false);
        }
        __syncthreads();
    }
#else
    // Fallback: direct-global fragment loads (f16, contiguous 32B chunks)
    for (int kk = 0; kk < DMODEL; kk += 32) {
        const v16h a = load_a_frag(val_h + (size_t)(m0 + wave * 16 + r) * DMODEL + kk, half);
        #pragma unroll
        for (int nt = 0; nt < 4; ++nt) {
            const v16h b = *(const v16h*)(Wz + (size_t)(n0 + nt * 16 + r) * DMODEL +
                                          kk + 16 * half);
            acc[nt] = __builtin_amdgcn_wmma_f32_16x16x32_f16(
                false, a, false, b, (short)0, acc[nt], false, false);
        }
    }
#endif

    // Epilogue: bias (+ q scale), store f16. q,k: [b,h][s][hd]; v: [b,h][hd][s].
    if (z == 2) {
        #pragma unroll
        for (int nt = 0; nt < 4; ++nt) {
            const int n = n0 + nt * 16 + r;
            const int h = n >> 6, d = n & 63;
            const float bn = bias[n];
            #pragma unroll
            for (int j = 0; j < 8; ++j) {
                const int m = m0 + wave * 16 + j + 8 * half;
                const int s = m >> 1, b = m & 1;
                vt[((size_t)(b * NH + h) * HDIM + d) * SLEN + s] =
                    (_Float16)(acc[nt][j] + bn);
            }
        }
    } else {
        _Float16* dst = (z == 0) ? qf : kf;
        const float qscale = (z == 0) ? 0.125f : 1.0f;  // 1/sqrt(hd)
        #pragma unroll
        for (int nt = 0; nt < 4; ++nt) {
            const int n = n0 + nt * 16 + r;
            const int h = n >> 6, d = n & 63;
            const float bn = bias[n];
            #pragma unroll
            for (int j = 0; j < 8; ++j) {
                const int m = m0 + wave * 16 + j + 8 * half;
                const int s = m >> 1, b = m & 1;
                dst[((size_t)(b * NH + h) * SLEN + s) * HDIM + d] =
                    (_Float16)((acc[nt][j] + bn) * qscale);
            }
        }
    }
}

// ---------------------------------------------------------------------------
// Kernel 2: causal sliding-window attention, flash-style online softmax.
// Block = 128 query rows of one (b,h); 8 waves x 16 rows. Key window:
// [q0-256, q0+127] = 12 tiles of 32 keys; valid keys for row r (in-block):
// t in [r, r+256] and absolute j >= 0, where t = j - (q0-256).
// ---------------------------------------------------------------------------
__global__ __launch_bounds__(256)
void attn_kernel(const _Float16* __restrict__ qf, const _Float16* __restrict__ kf,
                 const _Float16* __restrict__ vt, float* __restrict__ out)
{
    __shared__ _Float16 Plds[8 * 16 * 32];  // wave-private P tiles

    const int tid  = threadIdx.x;
    const int lane = tid & 31;
    const int wave = tid >> 5;
    const int half = lane >> 4;
    const int r    = lane & 15;

    const int x  = blockIdx.x;
    const int qb = x & 31;            // 32 query blocks of 128
    const int h  = (x >> 5) % NH;
    const int b  = x / (32 * NH);
    const int q0 = qb * 128;
    const int bh = b * NH + h;
    const int j0 = q0 - WSZ;

    const _Float16* qp = qf + (size_t)bh * SLEN * HDIM;
    const _Float16* kp = kf + (size_t)bh * SLEN * HDIM;
    const _Float16* vp = vt + (size_t)bh * HDIM * SLEN;

    const int rowbase = wave * 16;
    const float NEG_INF = -__builtin_inff();

    // Q fragments: 16 rows x 64, two K=32 chunks
    const _Float16* qbase = qp + (size_t)(q0 + rowbase + r) * HDIM;
    const v16h aq0 = load_a_frag(qbase, half);
    const v16h aq1 = load_a_frag(qbase + 32, half);

    v8f acc[4];
    float mrow[8], lrow[8];
    #pragma unroll
    for (int j = 0; j < 8; ++j) { mrow[j] = NEG_INF; lrow[j] = 0.0f; }
    #pragma unroll
    for (int nt = 0; nt < 4; ++nt)
        #pragma unroll
        for (int e = 0; e < 8; ++e) acc[nt][e] = 0.0f;

    _Float16* Pb = &Plds[wave * 16 * 32];

    for (int kt = 0; kt < 12; ++kt) {
        const int t0 = kt * 32;
        if (t0 + 31 < rowbase) continue;            // all keys before band
        if (t0 > rowbase + 15 + WSZ) continue;      // all keys after diagonal
        if (j0 + t0 + 31 < 0) continue;             // all keys before sequence

        // ---- scores: 16 rows x 32 keys ----
        v8f sc[2];
        #pragma unroll
        for (int nt = 0; nt < 2; ++nt)
            #pragma unroll
            for (int e = 0; e < 8; ++e) sc[nt][e] = 0.0f;

        #pragma unroll
        for (int nt = 0; nt < 2; ++nt) {
            int key = j0 + t0 + nt * 16 + r;
            if (key < 0) key = 0;  // clamped reads get masked below
            const _Float16* kb = kp + (size_t)key * HDIM + 16 * half;
            const v16h b0 = *(const v16h*)(kb);
            const v16h b1 = *(const v16h*)(kb + 32);
            sc[nt] = __builtin_amdgcn_wmma_f32_16x16x32_f16(
                false, aq0, false, b0, (short)0, sc[nt], false, false);
            sc[nt] = __builtin_amdgcn_wmma_f32_16x16x32_f16(
                false, aq1, false, b1, (short)0, sc[nt], false, false);
        }

        // ---- mask + online softmax update ----
        #pragma unroll
        for (int j = 0; j < 8; ++j) {
            const int rr = rowbase + j + 8 * half;  // C-layout row in block
            #pragma unroll
            for (int nt = 0; nt < 2; ++nt) {
                const int t   = t0 + nt * 16 + r;
                const int key = j0 + t;
                const bool valid = (key >= 0) && (t >= rr) && (t <= rr + WSZ);
                if (!valid) sc[nt][j] = NEG_INF;
            }
            float tm = fmaxf(sc[0][j], sc[1][j]);
            tm = fmaxf(tm, __shfl_xor(tm, 1, 32));
            tm = fmaxf(tm, __shfl_xor(tm, 2, 32));
            tm = fmaxf(tm, __shfl_xor(tm, 4, 32));
            tm = fmaxf(tm, __shfl_xor(tm, 8, 32));

            const float mn  = fmaxf(mrow[j], tm);
            const float scl = (mn == NEG_INF) ? 1.0f : __expf(mrow[j] - mn);
            const float p0  = (mn == NEG_INF) ? 0.0f : __expf(sc[0][j] - mn);
            const float p1  = (mn == NEG_INF) ? 0.0f : __expf(sc[1][j] - mn);

            float rs = p0 + p1;
            rs += __shfl_xor(rs, 1, 32);
            rs += __shfl_xor(rs, 2, 32);
            rs += __shfl_xor(rs, 4, 32);
            rs += __shfl_xor(rs, 8, 32);

            lrow[j] = lrow[j] * scl + rs;
            mrow[j] = mn;
            #pragma unroll
            for (int nt4 = 0; nt4 < 4; ++nt4) acc[nt4][j] *= scl;

            // stage P (f16) in wave-private LDS, row = C-layout row in tile
            Pb[(j + 8 * half) * 32 + r]      = (_Float16)p0;
            Pb[(j + 8 * half) * 32 + 16 + r] = (_Float16)p1;
        }

        // ---- O += P @ V_tile ----
        const v16h ap = load_a_frag(&Pb[r * 32], half);
        int kb2 = j0 + t0 + 16 * half;
        if (kb2 < 0) kb2 = 0;  // P is zero there
        #pragma unroll
        for (int nt4 = 0; nt4 < 4; ++nt4) {
            const int hd = nt4 * 16 + r;
            const v16h bvf = *(const v16h*)(vp + (size_t)hd * SLEN + kb2);
            acc[nt4] = __builtin_amdgcn_wmma_f32_16x16x32_f16(
                false, ap, false, bvf, (short)0, acc[nt4], false, false);
        }
    }

    // ---- normalize + store (S,B,D) f32 ----
    #pragma unroll
    for (int j = 0; j < 8; ++j) {
        const float inv = (lrow[j] > 0.0f) ? 1.0f / lrow[j] : 0.0f;
        const int rr = rowbase + j + 8 * half;
        const int s  = q0 + rr;
        #pragma unroll
        for (int nt4 = 0; nt4 < 4; ++nt4) {
            const int hd = nt4 * 16 + r;
            out[((size_t)s * BATCH + b) * DMODEL + h * HDIM + hd] = acc[nt4][j] * inv;
        }
    }
}

extern "C" void kernel_launch(void* const* d_in, const int* in_sizes, int n_in,
                              void* d_out, int out_size, void* d_ws, size_t ws_size,
                              hipStream_t stream) {
    const float* val = (const float*)d_in[0];
    const float* Wq  = (const float*)d_in[1];
    const float* bq  = (const float*)d_in[2];
    const float* Wk  = (const float*)d_in[3];
    const float* bk  = (const float*)d_in[4];
    const float* Wv  = (const float*)d_in[5];
    const float* bv  = (const float*)d_in[6];
    float* out = (float*)d_out;

    const size_t elems = (size_t)BATCH * NH * SLEN * HDIM;  // 6,291,456
    _Float16* qf    = (_Float16*)d_ws;
    _Float16* kf    = qf + elems;
    _Float16* vt    = kf + elems;
    _Float16* val_h = vt + elems;                            // 8192x768
    _Float16* Wt    = val_h + (size_t)MROWS * DMODEL;        // 3x768x768

    pack_val<<<dim3((MROWS * DMODEL) / (256 * 4)), 256, 0, stream>>>(val, val_h);
    transpose_w<<<dim3(DMODEL / 32, DMODEL / 32, 3), 256, 0, stream>>>(Wq, Wk, Wv, Wt);
    qkv_gemm<<<dim3(DMODEL / 64, MROWS / 128, 3), 256, 0, stream>>>(
        val_h, Wt, bq, bk, bv, qf, kf, vt);
    attn_kernel<<<dim3(BATCH * NH * (SLEN / 128)), 256, 0, stream>>>(qf, kf, vt, out);
}
